// DynamicNetworkAnt_30958124270370
// MI455X (gfx1250) — compile-verified, matrix operand order checked by reading, not run on hardware
//
#include <hip/hip_runtime.h>
#include <hip/hip_bf16.h>
#include <math.h>

// ---------------------------------------------------------------------------
// DynamicNetworkAnt forward for MI455X (gfx1250, wave32, WMMA).
// GEMMs: f16 WMMA, compile-time shapes, weights pre-packed into WMMA B-lane
// layout. Main GEMMs use multi-wave blocks with the A tile staged in LDS via
// global_load_async_to_lds_b128 (ASYNCcnt), double-buffered so async copies
// overlap WMMA issue. d_in pytree index map: JAX sorted-dict flatten order.
// ---------------------------------------------------------------------------

typedef __attribute__((ext_vector_type(16))) _Float16 v16h;
typedef __attribute__((ext_vector_type(8)))  float    v8f;

#define N_NODES 2048
#define N_EDGES 65536
#define HDIM    128
#define INV_SCALE 0.25f   // 1/sqrt(HEAD_D=16)

// ---------------- WMMA fragment helpers -------------------------------------

// A fragment: lane holds row r=lane&15, k in [kb,kb+8) and [kb+16,kb+24),
// kb=(lane>>4)*8. p points at row base + kb; works for global or LDS.
__device__ inline v16h load_a16(const float* p) {
  const float4* q = (const float4*)p;
  float4 x0 = q[0], x1 = q[1], x2 = q[4], x3 = q[5];
  v16h a;
  a[0] = (_Float16)x0.x;  a[1] = (_Float16)x0.y;  a[2] = (_Float16)x0.z;  a[3] = (_Float16)x0.w;
  a[4] = (_Float16)x1.x;  a[5] = (_Float16)x1.y;  a[6] = (_Float16)x1.z;  a[7] = (_Float16)x1.w;
  a[8] = (_Float16)x2.x;  a[9] = (_Float16)x2.y;  a[10] = (_Float16)x2.z; a[11] = (_Float16)x2.w;
  a[12] = (_Float16)x3.x; a[13] = (_Float16)x3.y; a[14] = (_Float16)x3.z; a[15] = (_Float16)x3.w;
  return a;
}

// Async copy of one 16B chunk global -> LDS (per-lane; ASYNCcnt-tracked).
__device__ inline void async_copy16(uint32_t lds_off, const float* g) {
  asm volatile("global_load_async_to_lds_b128 %0, %1, off"
               :: "v"(lds_off), "v"(g) : "memory");
}
__device__ inline void async_wait0() {
  asm volatile("s_wait_asynccnt 0" ::: "memory");
}

// Pack f32 weight W (KxN, or NxK if trans) into f16 WMMA B-lane layout:
// P[((kt*Nt + nt)*32 + lane)*16 + j], zero padded outside K/N.
__global__ void pack_b_kernel(const float* __restrict__ W, _Float16* __restrict__ P,
                              int K, int N, int trans) {
  int Kt = (K + 31) / 32, Nt = (N + 15) / 16;
  int total = Kt * Nt * 512;
  int idx = blockIdx.x * blockDim.x + threadIdx.x;
  if (idx >= total) return;
  int j    = idx & 15;
  int lane = (idx >> 4) & 31;
  int tile = idx >> 9;
  int nt = tile % Nt, kt = tile / Nt;
  int kb   = (lane >> 4) * 8;
  int koff = (j < 8) ? (kb + j) : (kb + 16 + (j - 8));
  int k = kt * 32 + koff;
  int n = nt * 16 + (lane & 15);
  float v = 0.0f;
  if (k < K && n < N) v = trans ? W[(size_t)n * K + k] : W[(size_t)k * N + n];
  P[idx] = (_Float16)v;
}

// ---------------------------------------------------------------------------
// LDS-staged GEMM: C[MxN] = act(A[MxK] @ W + bias).
// Block = NW waves (NW*32 threads), covers 32 rows x NW*32 cols.
// A tile (32x32 f32, 4KB) staged in LDS via async copies, double-buffered.
// K, N multiples of 32; M multiple of 32.
// ---------------------------------------------------------------------------
template <int K, int N, int NW, bool RELU>
__global__ void gemm_wmma_lds(const float* __restrict__ A,
                              const _Float16* __restrict__ PB,
                              const float* __restrict__ bias,
                              float* __restrict__ C) {
  constexpr int Kt  = K / 32;
  constexpr int Ntt = N / 16;
  constexpr int NT  = 2;                    // 16-col tiles per wave
  constexpr int MT  = 2;                    // 16-row tiles per block
  constexpr int CPT = 256 / (NW * 32);      // 16B chunks per thread per tile
  __shared__ float As[2][32 * 32];

  int tid  = threadIdx.x;
  int wave = tid >> 5;
  int lane = tid & 31;
  int m0 = blockIdx.y * 32;
  int n0 = blockIdx.x * (NW * 32) + wave * 32;
  int r = lane & 15, kb = (lane >> 4) << 3, half = lane >> 4;

  uint32_t lds_base = (uint32_t)(uintptr_t)&As[0][0];

  // per-thread chunk geometry is k-invariant: hoist pointers/offsets
  const float* gsrc[CPT];
  uint32_t     loff[CPT];
#pragma unroll
  for (int i = 0; i < CPT; ++i) {
    int c   = i * (NW * 32) + tid;          // 16B chunk id in [0,256)
    int row = c >> 3;
    int col = (c & 7) << 2;                 // float index within 32
    gsrc[i] = A + (size_t)(m0 + row) * K + col;
    loff[i] = (uint32_t)((row * 32 + col) * 4);
  }

  auto issue = [&](int kt, int buf) {
#pragma unroll
    for (int i = 0; i < CPT; ++i)
      async_copy16(lds_base + (uint32_t)(buf * 4096) + loff[i], gsrc[i] + kt * 32);
  };

  issue(0, 0);
  v8f acc[MT][NT] = {};
#pragma unroll
  for (int kt = 0; kt < Kt; ++kt) {
    int buf = kt & 1;
    async_wait0();
    __syncthreads();
    v16h a[MT];
#pragma unroll
    for (int mt = 0; mt < MT; ++mt)
      a[mt] = load_a16(&As[buf][(mt * 16 + r) * 32 + kb]);
    __syncthreads();
    if (kt + 1 < Kt) issue(kt + 1, buf ^ 1);   // overlap copies with WMMA
#pragma unroll
    for (int nt = 0; nt < NT; ++nt) {
      int ntg = (n0 >> 4) + nt;
      v16h b = *(const v16h*)(PB + ((size_t)(kt * Ntt + ntg) * 32 + lane) * 16);
#pragma unroll
      for (int mt = 0; mt < MT; ++mt)
        acc[mt][nt] = __builtin_amdgcn_wmma_f32_16x16x32_f16(
            false, a[mt], false, b, (short)0, acc[mt][nt], false, false);
    }
  }

#pragma unroll
  for (int mt = 0; mt < MT; ++mt)
#pragma unroll
    for (int nt = 0; nt < NT; ++nt) {
      int n = n0 + nt * 16 + r;
      float bv = bias[n];
#pragma unroll
      for (int v = 0; v < 8; ++v) {
        int mrow = m0 + mt * 16 + (half << 3) + v;
        float x = acc[mt][nt][v] + bv;
        if (RELU) x = fmaxf(x, 0.0f);
        C[(size_t)mrow * N + n] = x;
      }
    }
}

// Edge-head GEMM (K=320, N=128): A row e = [h[src[e]] | h[dst[e]] | emb[e]],
// gathered into LDS via async copies. Segment base pointers are k-invariant
// per thread and hoisted out of the k-loop (no index reloads in the hot loop).
__global__ void gemm_wmma_edge_lds(const float* __restrict__ h,
                                   const float* __restrict__ ee,
                                   const int* __restrict__ src,
                                   const int* __restrict__ dst,
                                   const _Float16* __restrict__ PB,
                                   const float* __restrict__ bias,
                                   float* __restrict__ C) {
  constexpr int K = 320, N = 128, NW = 4, NT = 2, MT = 2;
  constexpr int Kt = K / 32, Ntt = N / 16;
  constexpr int CPT = 256 / (NW * 32);
  __shared__ float As[2][32 * 32];

  int tid  = threadIdx.x;
  int wave = tid >> 5;
  int lane = tid & 31;
  int m0 = blockIdx.y * 32;
  int n0 = wave * 32;
  int r = lane & 15, kb = (lane >> 4) << 3, half = lane >> 4;

  uint32_t lds_base = (uint32_t)(uintptr_t)&As[0][0];

  // hoist per-chunk segment pointers (already offset by col) + LDS offsets
  const float* gsr[CPT];    // h[src[e]] + col
  const float* gds[CPT];    // h[dst[e]] + col
  const float* gem[CPT];    // emb[e]    + col
  uint32_t     loff[CPT];
#pragma unroll
  for (int i = 0; i < CPT; ++i) {
    int c   = i * (NW * 32) + tid;
    int row = c >> 3;
    int col = (c & 7) << 2;
    int e   = m0 + row;
    gsr[i] = h  + (size_t)src[e] * HDIM + col;
    gds[i] = h  + (size_t)dst[e] * HDIM + col;
    gem[i] = ee + (size_t)e * 64 + col;
    loff[i] = (uint32_t)((row * 32 + col) * 4);
  }

  auto issue = [&](int kt, int buf) {
    int k0 = kt * 32;     // folds under unroll; tiles never straddle segments
#pragma unroll
    for (int i = 0; i < CPT; ++i) {
      const float* g;
      if (k0 < 128)      g = gsr[i] + k0;
      else if (k0 < 256) g = gds[i] + (k0 - 128);
      else               g = gem[i] + (k0 - 256);
      async_copy16(lds_base + (uint32_t)(buf * 4096) + loff[i], g);
    }
  };

  issue(0, 0);
  v8f acc[MT][NT] = {};
#pragma unroll
  for (int kt = 0; kt < Kt; ++kt) {
    int buf = kt & 1;
    async_wait0();
    __syncthreads();
    v16h a[MT];
#pragma unroll
    for (int mt = 0; mt < MT; ++mt)
      a[mt] = load_a16(&As[buf][(mt * 16 + r) * 32 + kb]);
    __syncthreads();
    if (kt + 1 < Kt) issue(kt + 1, buf ^ 1);
#pragma unroll
    for (int nt = 0; nt < NT; ++nt) {
      int ntg = (n0 >> 4) + nt;
      v16h b = *(const v16h*)(PB + ((size_t)(kt * Ntt + ntg) * 32 + lane) * 16);
#pragma unroll
      for (int mt = 0; mt < MT; ++mt)
        acc[mt][nt] = __builtin_amdgcn_wmma_f32_16x16x32_f16(
            false, a[mt], false, b, (short)0, acc[mt][nt], false, false);
    }
  }

#pragma unroll
  for (int mt = 0; mt < MT; ++mt)
#pragma unroll
    for (int nt = 0; nt < NT; ++nt) {
      int n = n0 + nt * 16 + r;
      float bv = bias[n];
#pragma unroll
      for (int v = 0; v < 8; ++v) {
        int mrow = m0 + mt * 16 + (half << 3) + v;
        C[(size_t)mrow * N + n] = fmaxf(acc[mt][nt][v] + bv, 0.0f);
      }
    }
}

// Register-direct GEMM for the tiny N=10 classifier head.
template <int K, int N, int MT, int NT, bool RELU>
__global__ void gemm_wmma_t(const float* __restrict__ A,
                            const _Float16* __restrict__ PB,
                            const float* __restrict__ bias,
                            float* __restrict__ C) {
  constexpr int Kt = (K + 31) / 32;
  constexpr int Ntt = (N + 15) / 16;
  int lane = threadIdx.x;
  int n0 = blockIdx.x * (16 * NT);
  int m0 = blockIdx.y * (16 * MT);
  int r = lane & 15, kb = (lane >> 4) << 3;

  const float* arow[MT];
#pragma unroll
  for (int mt = 0; mt < MT; ++mt)
    arow[mt] = A + (size_t)(m0 + mt * 16 + r) * K + kb;

  v8f acc[MT][NT] = {};
#pragma unroll
  for (int kt = 0; kt < Kt; ++kt) {
    v16h a[MT];
#pragma unroll
    for (int mt = 0; mt < MT; ++mt) a[mt] = load_a16(arow[mt] + kt * 32);
#pragma unroll
    for (int nt = 0; nt < NT; ++nt) {
      int ntg = (n0 >> 4) + nt;
      v16h b = *(const v16h*)(PB + ((size_t)(kt * Ntt + ntg) * 32 + lane) * 16);
#pragma unroll
      for (int mt = 0; mt < MT; ++mt)
        acc[mt][nt] = __builtin_amdgcn_wmma_f32_16x16x32_f16(
            false, a[mt], false, b, (short)0, acc[mt][nt], false, false);
    }
  }

  int half = lane >> 4;
#pragma unroll
  for (int mt = 0; mt < MT; ++mt)
#pragma unroll
    for (int nt = 0; nt < NT; ++nt) {
      int n = n0 + nt * 16 + r;
      if ((N & 15) == 0 || n < N) {
        float bv = bias[n];
#pragma unroll
        for (int v = 0; v < 8; ++v) {
          int mrow = m0 + mt * 16 + (half << 3) + v;
          float x = acc[mt][nt][v] + bv;
          if (RELU) x = fmaxf(x, 0.0f);
          C[(size_t)mrow * N + n] = x;
        }
      }
    }
}

// ---------------------------- small VALU kernels ----------------------------

__global__ void zero_kernel(float* p, int n) {
  int i = blockIdx.x * blockDim.x + threadIdx.x;
  if (i < n) p[i] = 0.0f;
}

// pad edge_features 65536x16 -> 65536x32 (zeros beyond 16) for branch-free GEMM
__global__ void pad_ef_kernel(const float* __restrict__ EF, float* __restrict__ EFp) {
  int e = blockIdx.x, t = threadIdx.x;  // 32 threads
  EFp[(size_t)e * 32 + t] = (t < 16) ? EF[(size_t)e * 16 + t] : 0.0f;
}

// row LayerNorm in place; blockDim == D (64 or 128), one block per row.
__global__ void ln_kernel(float* __restrict__ x, const float* __restrict__ g,
                          const float* __restrict__ b) {
  __shared__ float red[128];
  int D = blockDim.x;
  int row = blockIdx.x, t = threadIdx.x;
  float v = x[(size_t)row * D + t];
  red[t] = v; __syncthreads();
  for (int s = D >> 1; s > 0; s >>= 1) { if (t < s) red[t] += red[t + s]; __syncthreads(); }
  float mean = red[0] / (float)D;
  __syncthreads();
  float d = v - mean;
  red[t] = d * d; __syncthreads();
  for (int s = D >> 1; s > 0; s >>= 1) { if (t < s) red[t] += red[t + s]; __syncthreads(); }
  float var = red[0] / (float)D;
  x[(size_t)row * D + t] = d * rsqrtf(var + 1e-5f) * g[t] + b[t];
}

// fold temporal conv1d(+time-mean) into windowed sums (U @ Wt + b == tmean)
__global__ void build_u_kernel(const float* __restrict__ t, float* __restrict__ U) {
  int n = blockIdx.x, c = threadIdx.x;  // c in [0,128)
  const float* tp = t + (size_t)n * 8 * HDIM + c;
  float vals[8], s_all = 0.0f;
#pragma unroll
  for (int s = 0; s < 8; ++s) { vals[s] = tp[s * HDIM]; s_all += vals[s]; }
  float s_first7 = s_all - vals[7];
  float s_last7  = s_all - vals[0];
  int de = c & ~1;
  float freq = __expf(-9.210340371976184f * (float)de / 128.0f);
  float ang  = (float)n * freq;
  float pe   = (c & 1) ? __cosf(ang) : __sinf(ang);
  size_t base = (size_t)n * 384 + (size_t)c * 3;
  U[base + 0] = (s_first7 + 7.0f * pe) * 0.125f;
  U[base + 1] = (s_all    + 8.0f * pe) * 0.125f;
  U[base + 2] = (s_last7  + 7.0f * pe) * 0.125f;
}

__global__ void scatter_add_kernel(const float* __restrict__ spatial,
                                   const int* __restrict__ src,
                                   const int* __restrict__ dst,
                                   float* __restrict__ agg) {
  int e = blockIdx.x, t = threadIdx.x;
  atomicAdd(&agg[(size_t)dst[e] * HDIM + t], spatial[(size_t)src[e] * HDIM + t]);
}

__global__ void concat2_kernel(const float* __restrict__ a, const float* __restrict__ a2,
                               const float* __restrict__ b, float* __restrict__ cat) {
  int n = blockIdx.x, t = threadIdx.x;  // 256
  float v;
  if (t < 128) { v = a[(size_t)n * 128 + t]; if (a2) v += a2[(size_t)n * 128 + t]; }
  else           v = b[(size_t)n * 128 + (t - 128)];
  cat[(size_t)n * 256 + t] = v;
}

// ---- sparse node attention (closed-form softmax over mostly-zero rows) -----

__global__ void edge_scores_kernel(const float* __restrict__ Q, const float* __restrict__ K,
                                   const int* __restrict__ src, const int* __restrict__ dst,
                                   float* __restrict__ sc) {
  int gid = blockIdx.x * blockDim.x + threadIdx.x;
  if (gid >= N_EDGES * 8) return;
  int e = gid >> 3, h = gid & 7;
  const float* q = Q + (size_t)src[e] * HDIM + h * 16;
  const float* k = K + (size_t)dst[e] * HDIM + h * 16;
  float s = 0.0f;
#pragma unroll
  for (int d = 0; d < 16; ++d) s += q[d] * k[d];
  sc[gid] = s * INV_SCALE;
}

__global__ void row_max_kernel(const float* __restrict__ sc,
                               const int* __restrict__ src, const int* __restrict__ dst,
                               float* __restrict__ m) {
  int gid = blockIdx.x * blockDim.x + threadIdx.x;
  if (gid >= N_EDGES * 8) return;
  int e = gid >> 3, h = gid & 7;
  float s = sc[gid];
  if (s > 0.0f) {  // implicit zeros => m >= 0; int atomicMax valid for s > 0
    atomicMax((int*)&m[src[e] * 8 + h], __float_as_int(s));
    atomicMax((int*)&m[dst[e] * 8 + h], __float_as_int(s));
  }
}

__global__ void z_init_kernel(const float* __restrict__ m, float* __restrict__ Z, int n) {
  int i = blockIdx.x * blockDim.x + threadIdx.x;
  if (i < n) Z[i] = (float)N_NODES * __expf(-m[i]);
}

__global__ void z_acc_kernel(const float* __restrict__ sc,
                             const int* __restrict__ src, const int* __restrict__ dst,
                             const float* __restrict__ m, float* __restrict__ Z) {
  int gid = blockIdx.x * blockDim.x + threadIdx.x;
  if (gid >= N_EDGES * 8) return;
  int e = gid >> 3, h = gid & 7;
  float s = sc[gid];
  float ms = m[src[e] * 8 + h], md = m[dst[e] * 8 + h];
  atomicAdd(&Z[src[e] * 8 + h], __expf(s - ms) - __expf(-ms));
  atomicAdd(&Z[dst[e] * 8 + h], __expf(s - md) - __expf(-md));
}

__global__ void colsum_kernel(const float* __restrict__ V, float* __restrict__ sumV) {
  __shared__ float red[256];
  int d = blockIdx.x, t = threadIdx.x;
  float s = 0.0f;
  for (int i = t; i < N_NODES; i += 256) s += V[(size_t)i * HDIM + d];
  red[t] = s; __syncthreads();
  for (int w = 128; w > 0; w >>= 1) { if (t < w) red[t] += red[t + w]; __syncthreads(); }
  if (t == 0) sumV[d] = red[0];
}

__global__ void attn_acc_kernel(const float* __restrict__ sc, const float* __restrict__ V,
                                const int* __restrict__ src, const int* __restrict__ dst,
                                const float* __restrict__ m, float* __restrict__ na) {
  int e = blockIdx.x, hd = threadIdx.x, h = hd >> 4;
  int s_ = src[e], d_ = dst[e];
  float s = sc[e * 8 + h];
  float ms = m[s_ * 8 + h], md = m[d_ * 8 + h];
  float ws = __expf(s - ms) - __expf(-ms);
  float wd = __expf(s - md) - __expf(-md);
  atomicAdd(&na[(size_t)s_ * HDIM + hd], ws * V[(size_t)d_ * HDIM + hd]);
  atomicAdd(&na[(size_t)d_ * HDIM + hd], wd * V[(size_t)s_ * HDIM + hd]);
}

__global__ void attn_final_kernel(float* __restrict__ na, const float* __restrict__ m,
                                  const float* __restrict__ Z, const float* __restrict__ sumV) {
  int n = blockIdx.x, hd = threadIdx.x, h = hd >> 4;
  float em = __expf(-m[n * 8 + h]);
  na[(size_t)n * HDIM + hd] = (na[(size_t)n * HDIM + hd] + em * sumV[hd]) / Z[n * 8 + h];
}

// ---------------------------- graph attention -------------------------------

__global__ void mean_cols_kernel(const float* __restrict__ h, float* __restrict__ gc) {
  int d = threadIdx.x;
  float s = 0.0f;
  for (int i = 0; i < N_NODES; ++i) s += h[(size_t)i * HDIM + d];
  gc[d] = s / (float)N_NODES;
}

__global__ void vecmat_kernel(const float* __restrict__ x, const float* __restrict__ W,
                              const float* __restrict__ b, float* __restrict__ y) {
  int j = threadIdx.x;
  float s = b[j];
  for (int i = 0; i < HDIM; ++i) s += x[i] * W[(size_t)i * HDIM + j];
  y[j] = s;
}

__global__ void gscore_kernel(const float* __restrict__ Qg, const float* __restrict__ Kv,
                              float* __restrict__ gsc) {
  int n = blockIdx.x * blockDim.x + threadIdx.x;
  if (n >= N_NODES) return;
  float s = 0.0f;
  for (int d = 0; d < HDIM; ++d) s += Qg[(size_t)n * HDIM + d] * Kv[d];
  gsc[n] = s * INV_SCALE;
}

__global__ void gsoftmax_stats_kernel(const float* __restrict__ gsc, float* __restrict__ stats) {
  __shared__ float red[256];
  int t = threadIdx.x;
  float mx = -1e30f;
  for (int i = t; i < N_NODES; i += 256) mx = fmaxf(mx, gsc[i]);
  red[t] = mx; __syncthreads();
  for (int w = 128; w > 0; w >>= 1) { if (t < w) red[t] = fmaxf(red[t], red[t + w]); __syncthreads(); }
  mx = red[0]; __syncthreads();
  float s = 0.0f;
  for (int i = t; i < N_NODES; i += 256) s += __expf(gsc[i] - mx);
  red[t] = s; __syncthreads();
  for (int w = 128; w > 0; w >>= 1) { if (t < w) red[t] += red[t + w]; __syncthreads(); }
  if (t == 0) { stats[0] = mx; stats[1] = red[0]; }
}

__global__ void ga_kernel(const float* __restrict__ gsc, const float* __restrict__ stats,
                          const float* __restrict__ Vv, float* __restrict__ ga) {
  int n = blockIdx.x, d = threadIdx.x;
  float w = __expf(gsc[n] - stats[0]) / stats[1];
  ga[(size_t)n * HDIM + d] = w * Vv[d];
}

__global__ void rowdot_kernel(const float* __restrict__ X, const float* __restrict__ w,
                              const float* __restrict__ bias, float* __restrict__ out,
                              int M, int D) {
  int n = blockIdx.x * blockDim.x + threadIdx.x;
  if (n >= M) return;
  float s = bias[0];
  for (int d = 0; d < D; ++d) s += X[(size_t)n * D + d] * w[d];
  out[n] = s;
}

// ---------------------------------------------------------------------------

extern "C" void kernel_launch(void* const* d_in, const int* in_sizes, int n_in,
                              void* d_out, int out_size, void* d_ws, size_t ws_size,
                              hipStream_t stream) {
  (void)in_sizes; (void)n_in; (void)out_size; (void)ws_size;
  const float* NF  = (const float*)d_in[0];
  const float* EF  = (const float*)d_in[1];
  const float* TNF = (const float*)d_in[2];
  const int*   EI  = (const int*)d_in[3];
  const int* src = EI;
  const int* dst = EI + N_EDGES;
  auto P = [&](int i) { return (const float*)d_in[i]; };

  float* ws = (float*)d_ws;
  size_t o = 0;
  auto alloc = [&](size_t n) {              // 32B-aligned slices
    n = (n + 7) & ~(size_t)7;
    float* p = ws + o; o += n; return p;
  };
  auto allocH = [&](size_t nh) { return (_Float16*)alloc((nh + 1) / 2); };

  float* h      = alloc((size_t)N_NODES * HDIM);
  float* h2     = alloc((size_t)N_NODES * HDIM);
  float* tbuf   = alloc((size_t)N_NODES * 8 * HDIM);
  float* U      = alloc((size_t)N_NODES * 384);
  float* EFp    = alloc((size_t)N_EDGES * 32);
  float* ebuf   = alloc((size_t)N_EDGES * 64);
  float* spatial= alloc((size_t)N_NODES * HDIM);
  float* agg    = alloc((size_t)N_NODES * HDIM);
  float* tmean  = alloc((size_t)N_NODES * HDIM);
  float* cat    = alloc((size_t)N_NODES * 256);
  float* Q      = alloc((size_t)N_NODES * HDIM);
  float* Km     = alloc((size_t)N_NODES * HDIM);
  float* V      = alloc((size_t)N_NODES * HDIM);
  float* Qg     = alloc((size_t)N_NODES * HDIM);
  float* sc     = alloc((size_t)N_EDGES * 8);
  float* m      = alloc((size_t)N_NODES * 8);
  float* Z      = alloc((size_t)N_NODES * 8);
  float* na     = alloc((size_t)N_NODES * HDIM);
  float* ga     = alloc((size_t)N_NODES * HDIM);
  float* sumV   = alloc(128);
  float* gc     = alloc(128);
  float* Kv     = alloc(128);
  float* Vv     = alloc(128);
  float* gsc    = alloc(N_NODES);
  float* stats  = alloc(8);
  float* cmid   = alloc((size_t)N_NODES * 64);
  float* emid   = alloc((size_t)N_EDGES * 128);
  float* dmid   = alloc((size_t)N_NODES * HDIM);

  _Float16* pb_ne  = allocH(2 * 8 * 512);   // (64,128)
  _Float16* pb_ee  = allocH(1 * 4 * 512);   // (16,64)
  _Float16* pb_sp  = allocH(4 * 8 * 512);   // (128,128)
  _Float16* pb_nq  = allocH(4 * 8 * 512);
  _Float16* pb_nk  = allocH(4 * 8 * 512);
  _Float16* pb_nv  = allocH(4 * 8 * 512);
  _Float16* pb_gq  = allocH(4 * 8 * 512);
  _Float16* pb_af  = allocH(8 * 8 * 512);   // (256,128)
  _Float16* pb_tf  = allocH(8 * 8 * 512);
  _Float16* pb_wt  = allocH(12 * 8 * 512);  // (384,128)
  _Float16* pb_s64 = allocH(4 * 4 * 512);   // (128,64) reused
  _Float16* pb_c2  = allocH(2 * 1 * 512);   // (64,10)
  _Float16* pb_d1  = allocH(4 * 8 * 512);
  _Float16* pb_d2  = allocH(4 * 8 * 512);
  _Float16* pb_e1  = allocH(10 * 8 * 512);  // (320,128)

  float* out = (float*)d_out;
  float* out_logits = out;          // 2048 x 10
  float* out_edge   = out + 20480;  // 65536
  float* out_cont   = out + 86016;  // 2048
  float* out_manip  = out + 88064;  // 2048
  float* out_dyn    = out + 90112;  // 2048 x 128

  dim3 b32(32), b64(64), b128(128);
  auto pack = [&](const float* W, _Float16* Pd, int K, int N, int trans) {
    int total = ((K + 31) / 32) * ((N + 15) / 16) * 512;
    pack_b_kernel<<<(total + 255) / 256, 256, 0, stream>>>(W, Pd, K, N, trans);
  };
  auto zero = [&](float* p, int n) {
    zero_kernel<<<(n + 255) / 256, 256, 0, stream>>>(p, n);
  };
  auto lnorm = [&](float* x, const float* g, const float* bb, int rows, int D) {
    ln_kernel<<<rows, D, 0, stream>>>(x, g, bb);
  };

  const int EH = N_EDGES * 8;

  // ---- encoders ----
  pack(P(121), pb_ne, 64, 128, 0);
  pack(P(23),  pb_ee, 16, 64, 0);
  pad_ef_kernel<<<N_EDGES, 32, 0, stream>>>(EF, EFp);

  gemm_wmma_lds<64, 128, 4, true><<<dim3(1, N_NODES / 32), b128, 0, stream>>>(NF, pb_ne, P(120), h);
  lnorm(h, P(123), P(122), N_NODES, 128);
  gemm_wmma_lds<64, 128, 4, true><<<dim3(1, N_NODES * 8 / 32), b128, 0, stream>>>(TNF, pb_ne, P(120), tbuf);
  lnorm(tbuf, P(123), P(122), N_NODES * 8, 128);
  build_u_kernel<<<N_NODES, 128, 0, stream>>>(tbuf, U);
  gemm_wmma_lds<32, 64, 2, true><<<dim3(1, N_EDGES / 32), b64, 0, stream>>>(EFp, pb_ee, P(22), ebuf);
  lnorm(ebuf, P(25), P(24), N_EDGES, 64);

  // ---- layers ----
  for (int L = 0; L < 4; ++L) {
    const int LB = 28 + 22 * L;
    const float *af_b = P(LB+0),  *af_w = P(LB+1);
    const float *gk_b = P(LB+2),  *gk_w = P(LB+3);
    const float *gq_b = P(LB+4),  *gq_w = P(LB+5);
    const float *gv_b = P(LB+6),  *gv_w = P(LB+7);
    const float *nk_b = P(LB+8),  *nk_w = P(LB+9);
    const float *nq_b = P(LB+10), *nq_w = P(LB+11);
    const float *nv_b = P(LB+12), *nv_w = P(LB+13);
    const float *tf_b = P(LB+14), *tf_w = P(LB+15);
    const float *tfl_b= P(LB+16), *tfl_g= P(LB+17);
    const float *sp_b = P(LB+18), *sp_w = P(LB+19);
    const float *tc_b = P(LB+20), *tc_w = P(LB+21);

    pack(sp_w, pb_sp, 128, 128, 0);
    pack(nq_w, pb_nq, 128, 128, 0);
    pack(nk_w, pb_nk, 128, 128, 0);
    pack(nv_w, pb_nv, 128, 128, 0);
    pack(gq_w, pb_gq, 128, 128, 0);
    pack(af_w, pb_af, 256, 128, 0);
    pack(tf_w, pb_tf, 256, 128, 0);
    pack(tc_w, pb_wt, 384, 128, 1);  // transpose: Wt[i*3+t][o] = w[o][i*3+t]

    // TGC
    gemm_wmma_lds<128, 128, 4, false><<<dim3(1, 64), b128, 0, stream>>>(h, pb_sp, sp_b, spatial);
    zero(agg, N_NODES * HDIM);
    scatter_add_kernel<<<N_EDGES, 128, 0, stream>>>(spatial, src, dst, agg);
    gemm_wmma_lds<384, 128, 4, false><<<dim3(1, 64), b128, 0, stream>>>(U, pb_wt, tc_b, tmean);
    concat2_kernel<<<N_NODES, 256, 0, stream>>>(spatial, agg, tmean, cat);
    gemm_wmma_lds<256, 128, 4, true><<<dim3(1, 64), b128, 0, stream>>>(cat, pb_tf, tf_b, h2);
    lnorm(h2, tfl_g, tfl_b, N_NODES, 128);

    // node attention (sparse closed-form softmax)
    gemm_wmma_lds<128, 128, 4, false><<<dim3(1, 64), b128, 0, stream>>>(h2, pb_nq, nq_b, Q);
    gemm_wmma_lds<128, 128, 4, false><<<dim3(1, 64), b128, 0, stream>>>(h2, pb_nk, nk_b, Km);
    gemm_wmma_lds<128, 128, 4, false><<<dim3(1, 64), b128, 0, stream>>>(h2, pb_nv, nv_b, V);
    edge_scores_kernel<<<(EH + 255) / 256, 256, 0, stream>>>(Q, Km, src, dst, sc);
    zero(m, N_NODES * 8);
    row_max_kernel<<<(EH + 255) / 256, 256, 0, stream>>>(sc, src, dst, m);
    z_init_kernel<<<(N_NODES * 8 + 255) / 256, 256, 0, stream>>>(m, Z, N_NODES * 8);
    z_acc_kernel<<<(EH + 255) / 256, 256, 0, stream>>>(sc, src, dst, m, Z);
    colsum_kernel<<<128, 256, 0, stream>>>(V, sumV);
    zero(na, N_NODES * HDIM);
    attn_acc_kernel<<<N_EDGES, 128, 0, stream>>>(sc, V, src, dst, m, na);
    attn_final_kernel<<<N_NODES, 128, 0, stream>>>(na, m, Z, sumV);

    // graph attention
    mean_cols_kernel<<<1, 128, 0, stream>>>(h2, gc);
    vecmat_kernel<<<1, 128, 0, stream>>>(gc, gk_w, gk_b, Kv);
    vecmat_kernel<<<1, 128, 0, stream>>>(gc, gv_w, gv_b, Vv);
    gemm_wmma_lds<128, 128, 4, false><<<dim3(1, 64), b128, 0, stream>>>(h2, pb_gq, gq_b, Qg);
    gscore_kernel<<<(N_NODES + 255) / 256, 256, 0, stream>>>(Qg, Kv, gsc);
    gsoftmax_stats_kernel<<<1, 256, 0, stream>>>(gsc, stats);
    ga_kernel<<<N_NODES, 128, 0, stream>>>(gsc, stats, Vv, ga);

    // fusion -> h for next layer
    concat2_kernel<<<N_NODES, 256, 0, stream>>>(na, nullptr, ga, cat);
    gemm_wmma_lds<256, 128, 4, true><<<dim3(1, 64), b128, 0, stream>>>(cat, pb_af, af_b, h);
  }

  // ---- heads ----
  // node classification
  pack(P(5), pb_s64, 128, 64, 0);
  gemm_wmma_lds<128, 64, 2, true><<<dim3(1, 64), b64, 0, stream>>>(h, pb_s64, P(4), cmid);
  lnorm(cmid, P(9), P(8), N_NODES, 64);
  pack(P(7), pb_c2, 64, 10, 0);
  gemm_wmma_t<64, 10, 2, 1, false><<<dim3(1, 64), b32, 0, stream>>>(cmid, pb_c2, P(6), out_logits);

  // edge prediction (gathered A rows, K=320)
  pack(P(19), pb_e1, 320, 128, 0);
  gemm_wmma_edge_lds<<<dim3(1, N_EDGES / 32), b128, 0, stream>>>(h, ebuf, src, dst, pb_e1, P(18), emid);
  lnorm(emid, P(27), P(26), N_EDGES, 128);
  rowdot_kernel<<<(N_EDGES + 255) / 256, 256, 0, stream>>>(emid, P(21), P(20), out_edge, N_EDGES, 128);

  // contagion
  pack(P(11), pb_s64, 128, 64, 0);
  gemm_wmma_lds<128, 64, 2, true><<<dim3(1, 64), b64, 0, stream>>>(h, pb_s64, P(10), cmid);
  rowdot_kernel<<<(N_NODES + 255) / 256, 256, 0, stream>>>(cmid, P(13), P(12), out_cont, N_NODES, 64);

  // manipulation
  pack(P(117), pb_s64, 128, 64, 0);
  gemm_wmma_lds<128, 64, 2, true><<<dim3(1, 64), b64, 0, stream>>>(h, pb_s64, P(116), cmid);
  rowdot_kernel<<<(N_NODES + 255) / 256, 256, 0, stream>>>(cmid, P(119), P(118), out_manip, N_NODES, 64);

  // dynamics
  pack(P(15), pb_d1, 128, 128, 0);
  gemm_wmma_lds<128, 128, 4, true><<<dim3(1, 64), b128, 0, stream>>>(h, pb_d1, P(14), dmid);
  pack(P(17), pb_d2, 128, 128, 0);
  gemm_wmma_lds<128, 128, 4, false><<<dim3(1, 64), b128, 0, stream>>>(dmid, pb_d2, P(16), out_dyn);
}
